// ContinuousSpatialSSM_V2_78340203479243
// MI455X (gfx1250) — matrix operand order repaired and checked by reading, not compiled, not run on hardware
//
#include <hip/hip_runtime.h>
#include <hip/hip_bf16.h>

typedef __attribute__((ext_vector_type(16))) _Float16 v16h;
typedef __attribute__((ext_vector_type(8)))  float    v8f;

#define DMODEL 384
#define NTOK   1024
#define BSZ    4
#define NST    16
#define MTOT   (BSZ * NTOK)   // 4096

__device__ __forceinline__ float softplus_f(float v) {
    return (v > 20.0f) ? v : log1pf(expf(v));
}

// ---------------------------------------------------------------------------
// Kernel 1: WMMA GEMM + fused softplus/min epilogue.
// Out[m][n] = min(softplus( sum_k X[m][k] * W[n][k] + bias[n] ), 0.15)
// X: (M x K) row-major, W: (N x K) row-major. Requires M%64==0, N%64==0, K%32==0.
// Block = 256 threads (8 waves). Block tile 64x64; wave tile 16x32 (2 WMMA).
// ---------------------------------------------------------------------------
__global__ __launch_bounds__(256)
void gemm_dt_kernel(const float* __restrict__ X,
                    const float* __restrict__ W,
                    const float* __restrict__ bias,
                    float* __restrict__ Out,
                    int M, int N, int K)
{
    __shared__ _Float16 sA[64][40];   // X tile, f16, padded rows
    __shared__ _Float16 sB[64][40];   // W tile, f16

    const int tid  = threadIdx.x;
    const int lane = tid & 31;
    const int wave = tid >> 5;
    const int half = lane >> 4;          // 0 or 1
    const int l16  = lane & 15;
    const int msub = (wave & 3) << 4;    // 0,16,32,48
    const int nsub = (wave >> 2) << 5;   // 0,32

    const int m0 = blockIdx.y * 64;
    const int n0 = blockIdx.x * 64;

    // staging: each thread loads 8 contiguous floats of one tile row
    const int lr = tid >> 2;          // 0..63
    const int lc = (tid & 3) * 8;     // 0,8,16,24

    v8f c0 = {};
    v8f c1 = {};

    for (int k0 = 0; k0 < K; k0 += 32) {
        const float4* xa = (const float4*)(X + (size_t)(m0 + lr) * K + k0 + lc);
        const float4* wa = (const float4*)(W + (size_t)(n0 + lr) * K + k0 + lc);
        float4 x0 = xa[0], x1 = xa[1];
        float4 w0 = wa[0], w1 = wa[1];
        __syncthreads();   // protect previous iteration's fragment reads
        sA[lr][lc + 0] = (_Float16)x0.x; sA[lr][lc + 1] = (_Float16)x0.y;
        sA[lr][lc + 2] = (_Float16)x0.z; sA[lr][lc + 3] = (_Float16)x0.w;
        sA[lr][lc + 4] = (_Float16)x1.x; sA[lr][lc + 5] = (_Float16)x1.y;
        sA[lr][lc + 6] = (_Float16)x1.z; sA[lr][lc + 7] = (_Float16)x1.w;
        sB[lr][lc + 0] = (_Float16)w0.x; sB[lr][lc + 1] = (_Float16)w0.y;
        sB[lr][lc + 2] = (_Float16)w0.z; sB[lr][lc + 3] = (_Float16)w0.w;
        sB[lr][lc + 4] = (_Float16)w1.x; sB[lr][lc + 5] = (_Float16)w1.y;
        sB[lr][lc + 6] = (_Float16)w1.z; sB[lr][lc + 7] = (_Float16)w1.w;
        __syncthreads();

        // A fragment: lane holds row m = msub+l16.
        // halves 0..7  -> K = half*8 + i ; halves 8..15 -> K = 16 + half*8 + i
        v16h a;
        const int am = msub + l16;
        #pragma unroll
        for (int i = 0; i < 8; ++i) {
            a[i]     = sA[am][half * 8 + i];
            a[i + 8] = sA[am][16 + half * 8 + i];
        }
        // B fragments: lane holds column n; b[i] -> K = (i>>3)*16 + half*8 + (i&7)
        v16h b0, b1;
        const int bn0 = nsub + l16;
        #pragma unroll
        for (int i = 0; i < 16; ++i) {
            b0[i] = sB[bn0][(i >> 3) * 16 + half * 8 + (i & 7)];
            b1[i] = sB[bn0 + 16][(i >> 3) * 16 + half * 8 + (i & 7)];
        }

        c0 = __builtin_amdgcn_wmma_f32_16x16x32_f16(false, a, false, b0,
                                                    (short)0, c0, false, false);
        c1 = __builtin_amdgcn_wmma_f32_16x16x32_f16(false, a, false, b1,
                                                    (short)0, c1, false, false);
    }

    // Epilogue. D layout: lanes 0-15 -> M = r, lanes 16-31 -> M = 8+r.
    const int mbase = m0 + msub + (half << 3);
    const int ng0   = n0 + nsub + l16;
    const float bia0 = bias[ng0];
    const float bia1 = bias[ng0 + 16];
    #pragma unroll
    for (int r = 0; r < 8; ++r) {
        float v0 = fminf(softplus_f(c0[r] + bia0), 0.15f);
        float v1 = fminf(softplus_f(c1[r] + bia1), 0.15f);
        Out[(size_t)(mbase + r) * N + ng0]      = v0;
        Out[(size_t)(mbase + r) * N + ng0 + 16] = v1;
    }
}

// ---------------------------------------------------------------------------
// Kernel 2: skinny projection Out[m][s] = sum_k X[m][k] * W[s][k]
// X: (4096 x 384), W: (16 x 384), Out: (4096 x 16). Exact fp32.
// ---------------------------------------------------------------------------
__global__ __launch_bounds__(256)
void proj16_kernel(const float* __restrict__ X,
                   const float* __restrict__ W,
                   float* __restrict__ Out)
{
    const int gid = blockIdx.x * 256 + threadIdx.x;
    const int m = gid >> 4;
    const int s = gid & 15;
    const float4* xr = (const float4*)(X + (size_t)m * DMODEL);
    const float4* wr = (const float4*)(W + (size_t)s * DMODEL);
    float acc = 0.0f;
    #pragma unroll 4
    for (int k = 0; k < DMODEL / 4; ++k) {
        float4 av = xr[k], wv = wr[k];
        acc += av.x * wv.x + av.y * wv.y + av.z * wv.z + av.w * wv.w;
    }
    Out[(size_t)m * NST + s] = acc;
}

// ---------------------------------------------------------------------------
// Kernel 3: per-(b,d) spatial SSM. Entire h (1024 x 16) lives in LDS across all
// K diffusion steps; layout [s][n]. Stencil phase: each thread owns 4
// CONSECUTIVE tokens -> per state: 3x ds_load_b128 (center/up/down) +
// 2x ds_load_b32 (left/right), interior horizontal neighbors reused from the
// float4 in registers. Updates staged in registers (float4 rnew[16]).
// grid = (384, 4), block = 256 threads.
// ---------------------------------------------------------------------------
__global__ __launch_bounds__(256)
void ssm_spatial_kernel(const float* __restrict__ x,      // (4,1024,384)
                        const float* __restrict__ ds_g,   // (4,1024,384)
                        const float* __restrict__ dd_g,   // (4,1024,384)
                        const float* __restrict__ Bm,     // (4,1024,16)
                        const float* __restrict__ Cm,     // (4,1024,16)
                        const float* __restrict__ A_log,  // (384,16)
                        const float* __restrict__ diff_raw, // (384)
                        const float* __restrict__ D_param,  // (384)
                        const int*   __restrict__ Kp,       // scalar
                        float* __restrict__ y)              // (4,1024,384)
{
    __shared__ float sh_h[NST][NTOK];   // 64 KB
    __shared__ float sh_B[NST][NTOK];   // 64 KB
    __shared__ float sh_x[NTOK];
    __shared__ float sh_ds[NTOK];
    __shared__ float sh_dd[NTOK];
    __shared__ float sh_A[NST];

    const int tid = threadIdx.x;
    const int d = blockIdx.x;
    const int b = blockIdx.y;
    const size_t colbase = ((size_t)b * NTOK) * DMODEL + d;

    // -------- load phase (strided-token mapping, conflict-free b32 writes) --
    #pragma unroll
    for (int j = 0; j < 4; ++j) {
        const int n = tid + 256 * j;
        sh_x[n]  = x[colbase + (size_t)n * DMODEL];
        sh_ds[n] = ds_g[colbase + (size_t)n * DMODEL];
        sh_dd[n] = dd_g[colbase + (size_t)n * DMODEL];
    }
    if (tid < NST) sh_A[tid] = -softplus_f(A_log[d * NST + tid]);
    const float Dc = 0.5f / (1.0f + expf(-diff_raw[d]));

    // B_mat -> LDS transposed to [s][n]
    const float4* B4 = (const float4*)(Bm + (size_t)b * NTOK * NST);
    #pragma unroll
    for (int j = 0; j < 4; ++j) {
        const int n = tid + 256 * j;
        float q[NST];
        float4 q0 = B4[n * 4 + 0], q1 = B4[n * 4 + 1];
        float4 q2 = B4[n * 4 + 2], q3 = B4[n * 4 + 3];
        q[0]=q0.x; q[1]=q0.y; q[2]=q0.z; q[3]=q0.w;
        q[4]=q1.x; q[5]=q1.y; q[6]=q1.z; q[7]=q1.w;
        q[8]=q2.x; q[9]=q2.y; q[10]=q2.z; q[11]=q2.w;
        q[12]=q3.x; q[13]=q3.y; q[14]=q3.z; q[15]=q3.w;
        #pragma unroll
        for (int s = 0; s < NST; ++s) sh_B[s][n] = q[s];
    }
    __syncthreads();

    // -------- stencil phase: 4 consecutive tokens per thread ----------------
    const int n0  = tid * 4;                   // all 4 tokens in one grid row
    const int row = n0 >> 5;
    const int cc  = n0 & 31;                   // 0,4,...,28
    const int nu0 = (row == 0)  ? n0 : n0 - 32;   // replicate padding
    const int nd0 = (row == 31) ? n0 : n0 + 32;
    const int nl  = (cc == 0)   ? n0 : n0 - 1;    // left  of token n0
    const int nr  = (cc == 28)  ? n0 + 3 : n0 + 4; // right of token n0+3

    const float4 x4  = *(const float4*)&sh_x[n0];
    const float4 dsv = *(const float4*)&sh_ds[n0];
    const float4 ddv = *(const float4*)&sh_dd[n0];

    // init h = h0 = x (outer) B
    #pragma unroll
    for (int s = 0; s < NST; ++s) {
        float4 bv = *(const float4*)&sh_B[s][n0];
        float4 hv;
        hv.x = x4.x * bv.x; hv.y = x4.y * bv.y;
        hv.z = x4.z * bv.z; hv.w = x4.w * bv.w;
        *(float4*)&sh_h[s][n0] = hv;
    }
    __syncthreads();

    const int Ks = Kp[0];
    const float dtv = 1.0f / (float)Ks;
    float4 as4, ad4;
    as4.x = dsv.x * dtv; as4.y = dsv.y * dtv;
    as4.z = dsv.z * dtv; as4.w = dsv.w * dtv;
    const float dcdt = Dc * dtv;
    ad4.x = ddv.x * dcdt; ad4.y = ddv.y * dcdt;
    ad4.z = ddv.z * dcdt; ad4.w = ddv.w * dcdt;

    for (int k = 0; k < Ks; ++k) {
        float4 rnew[NST];
        #pragma unroll
        for (int s = 0; s < NST; ++s) {
            const float* hrow = &sh_h[s][0];
            float4 hc = *(const float4*)&hrow[n0];
            float4 hu = *(const float4*)&hrow[nu0];
            float4 hd = *(const float4*)&hrow[nd0];
            float  hl = hrow[nl];
            float  hr = hrow[nr];
            float4 bv = *(const float4*)&sh_B[s][n0];
            const float As = sh_A[s];
            float lap_x = hu.x + hd.x + hl   + hc.y - 4.0f * hc.x;
            float lap_y = hu.y + hd.y + hc.x + hc.z - 4.0f * hc.y;
            float lap_z = hu.z + hd.z + hc.y + hc.w - 4.0f * hc.z;
            float lap_w = hu.w + hd.w + hc.z + hr   - 4.0f * hc.w;
            rnew[s].x = hc.x + as4.x * (As * hc.x + x4.x * bv.x) + ad4.x * lap_x;
            rnew[s].y = hc.y + as4.y * (As * hc.y + x4.y * bv.y) + ad4.y * lap_y;
            rnew[s].z = hc.z + as4.z * (As * hc.z + x4.z * bv.z) + ad4.z * lap_z;
            rnew[s].w = hc.w + as4.w * (As * hc.w + x4.w * bv.w) + ad4.w * lap_w;
        }
        __syncthreads();
        #pragma unroll
        for (int s = 0; s < NST; ++s) *(float4*)&sh_h[s][n0] = rnew[s];
        __syncthreads();
    }

    // -------- contraction: y = sum_s h*C + x*D (strided-token mapping) ------
    const float4* C4 = (const float4*)(Cm + (size_t)b * NTOK * NST);
    const float Dp = D_param[d];
    #pragma unroll
    for (int j = 0; j < 4; ++j) {
        const int n = tid + 256 * j;
        float q[NST];
        float4 q0 = C4[n * 4 + 0], q1 = C4[n * 4 + 1];
        float4 q2 = C4[n * 4 + 2], q3 = C4[n * 4 + 3];
        q[0]=q0.x; q[1]=q0.y; q[2]=q0.z; q[3]=q0.w;
        q[4]=q1.x; q[5]=q1.y; q[6]=q1.z; q[7]=q1.w;
        q[8]=q2.x; q[9]=q2.y; q[10]=q2.z; q[11]=q2.w;
        q[12]=q3.x; q[13]=q3.y; q[14]=q3.z; q[15]=q3.w;
        float acc = 0.0f;
        #pragma unroll
        for (int s = 0; s < NST; ++s) acc += sh_h[s][n] * q[s];
        y[colbase + (size_t)n * DMODEL] = acc + sh_x[n] * Dp;
    }
}

// ---------------------------------------------------------------------------
extern "C" void kernel_launch(void* const* d_in, const int* in_sizes, int n_in,
                              void* d_out, int out_size, void* d_ws, size_t ws_size,
                              hipStream_t stream) {
    (void)in_sizes; (void)n_in; (void)out_size; (void)ws_size;

    const float* x        = (const float*)d_in[0];   // (4,1024,384)
    const float* W_dts    = (const float*)d_in[1];   // (384,384)
    const float* b_dts    = (const float*)d_in[2];   // (384)
    const float* W_dtd    = (const float*)d_in[3];   // (384,384)
    const float* b_dtd    = (const float*)d_in[4];   // (384)
    const float* W_B      = (const float*)d_in[5];   // (16,384)
    const float* W_C      = (const float*)d_in[6];   // (16,384)
    const float* D_param  = (const float*)d_in[7];   // (384)
    const float* A_log    = (const float*)d_in[8];   // (384,16)
    const float* diff_raw = (const float*)d_in[9];   // (384)
    const int*   K_steps  = (const int*)d_in[10];    // scalar
    float* y = (float*)d_out;

    float* ws = (float*)d_ws;
    float* ds = ws;                                  // 4096*384
    float* dd = ds + (size_t)MTOT * DMODEL;          // 4096*384
    float* Bm = dd + (size_t)MTOT * DMODEL;          // 4096*16
    float* Cm = Bm + (size_t)MTOT * NST;             // 4096*16

    // delta GEMMs (WMMA, fused softplus/min)
    dim3 gBlk(256);
    dim3 gGrid(DMODEL / 64, MTOT / 64);              // (6, 64)
    gemm_dt_kernel<<<gGrid, gBlk, 0, stream>>>(x, W_dts, b_dts, ds, MTOT, DMODEL, DMODEL);
    gemm_dt_kernel<<<gGrid, gBlk, 0, stream>>>(x, W_dtd, b_dtd, dd, MTOT, DMODEL, DMODEL);

    // B/C projections
    dim3 pGrid((MTOT * NST) / 256);                  // 256 blocks
    proj16_kernel<<<pGrid, gBlk, 0, stream>>>(x, W_B, Bm);
    proj16_kernel<<<pGrid, gBlk, 0, stream>>>(x, W_C, Cm);

    // fused spatial SSM: one workgroup per (b, d) channel, h resident in LDS
    dim3 sGrid(DMODEL, BSZ);                         // (384, 4)
    ssm_spatial_kernel<<<sGrid, gBlk, 0, stream>>>(x, ds, dd, Bm, Cm,
                                                   A_log, diff_raw, D_param,
                                                   K_steps, y);
}
